// CrossAttention_33380485825043
// MI455X (gfx1250) — compile-verified, matrix operand order checked by reading, not run on hardware
//
#include <hip/hip_runtime.h>
#include <hip/hip_bf16.h>

// ---------------- problem constants ----------------
#define BATCH 4
#define SQ    4096
#define SKV   1024
#define EMB   1024
#define CDIM  768
#define HEADS 16
#define DHEAD 64

typedef __attribute__((ext_vector_type(16))) _Float16 v16h;
typedef __attribute__((ext_vector_type(8)))  _Float16 v8h;
typedef __attribute__((ext_vector_type(8)))  float    v8f;
typedef __attribute__((ext_vector_type(4)))  float    v4f;
typedef __attribute__((ext_vector_type(4)))  unsigned int v4u;
typedef __attribute__((ext_vector_type(8)))  int      v8i;
typedef __attribute__((ext_vector_type(4)))  int      v4i;

// ---------------------------------------------------------------------------
// CDNA5 WMMA 16x16x32 f16 -> f32.  A/B fragment layout (ISA 7.12.2, 16-bit):
//   lane L: half = L>>4, r = L&15
//   element i of v16h  <->  K = (i<8) ? 8*half+i : 16+8*half+(i-8)
//   A: lane holds row M=r ; B: lane holds column N=r
// ---------------------------------------------------------------------------
__device__ __forceinline__ v8f wmma_f16(v16h a, v16h b, v8f c) {
    return __builtin_amdgcn_wmma_f32_16x16x32_f16(
        /*neg_a=*/false, a, /*neg_b=*/false, b,
        /*c_mod=*/(short)0, c, /*reuse_a=*/false, /*reuse_b=*/false);
}

// load one 16-element fragment (one row/col, K-chunk of 32) from f16 memory
__device__ __forceinline__ v16h frag_h16(const _Float16* p, int half) {
    const _Float16* p1 = p + 8 * half;          // K = 8h .. 8h+7
    v8h x = *(const v8h*)p1;
    v8h y = *(const v8h*)(p1 + 16);             // K = 16+8h .. 16+8h+7
    v16h f;
#pragma unroll
    for (int i = 0; i < 8; ++i) { f[i] = x[i]; f[8 + i] = y[i]; }
    return f;
}

// same but source is fp32 (convert to f16 in registers)
__device__ __forceinline__ v16h frag_f32(const float* __restrict__ p, int half) {
    const float* p1 = p + 8 * half;
    v4f a0 = *(const v4f*)p1;
    v4f a1 = *(const v4f*)(p1 + 4);
    v4f b0 = *(const v4f*)(p1 + 16);
    v4f b1 = *(const v4f*)(p1 + 20);
    v16h f;
#pragma unroll
    for (int i = 0; i < 4; ++i) {
        f[i]      = (_Float16)a0[i];
        f[4 + i]  = (_Float16)a1[i];
        f[8 + i]  = (_Float16)b0[i];
        f[12 + i] = (_Float16)b1[i];
    }
    return f;
}

// ---------------------------------------------------------------------------
// Tensor Data Mover: 2D f16 tile (dim1 rows x dim0 elems) global -> LDS.
// HW padding: +4 dwords (8 f16) after every 32 dwords (64 f16 = one row) so
// LDS row stride = 72 f16 = 144 B (16B-aligned, bank-conflict-free reads).
// D# layout per CDNA5 ISA; clang-23 6-arg builtin
// (group0 v4u, group1 v8i, group2 v4i, group3 v4i, extra v8i zero, cpol).
// ---------------------------------------------------------------------------
#define LDSROW 72   // f16 elements per LDS row after TDM padding

__device__ __forceinline__ void tdm_load_2d_f16(unsigned int lds_off,
                                                const void* gaddr,
                                                unsigned int dim0,      // elems/row
                                                unsigned int dim1,      // rows
                                                unsigned int stride0) { // elems
    unsigned long long ga = (unsigned long long)(__UINTPTR_TYPE__)gaddr;
    v4u g0;
    g0.x = 1u;                                            // count=1 (valid, user)
    g0.y = lds_off;                                       // lds_addr
    g0.z = (unsigned int)ga;                              // global_addr[31:0]
    g0.w = ((unsigned int)(ga >> 32) & 0x01FFFFFFu)       // global_addr[56:32]
           | (2u << 30);                                  // type=2 ("image")
    v8i g1;
    g1[0] = (int)((1u << 16)                              // data_size = 2 bytes
                  | (1u << 20)                            // pad_enable
                  | (4u << 22)                            // pad_interval: 32 dw
                  | (3u << 25));                          // pad_amount: 4 dw
    g1[1] = (int)((dim0 & 0xFFFFu) << 16);                // tensor_dim0[15:0]
    g1[2] = (int)(((dim0 >> 16) & 0xFFFFu) | ((dim1 & 0xFFFFu) << 16));
    g1[3] = (int)(((dim1 >> 16) & 0xFFFFu) | ((dim0 & 0xFFFFu) << 16)); // tile_dim0
    g1[4] = (int)(dim1 & 0xFFFFu);                        // tile_dim1 (tile_dim2=0)
    g1[5] = (int)stride0;                                 // tensor_dim0_stride lo
    g1[6] = 0;
    g1[7] = 0;
    v4i g2 = {0, 0, 0, 0};
    v4i g3 = {0, 0, 0, 0};
    v8i g4 = {0, 0, 0, 0, 0, 0, 0, 0};
    __builtin_amdgcn_tensor_load_to_lds(g0, g1, g2, g3, g4, 0);
}

// ---------------------------------------------------------------------------
// W[K][N] fp32 -> Wt[N][K] f16   (so GEMM B-fragments are contiguous b128)
// ---------------------------------------------------------------------------
__global__ __launch_bounds__(256)
void xattn_transpose_convert(const float* __restrict__ W, _Float16* __restrict__ Wt,
                             int K, int N) {
    __shared__ _Float16 tile[64][72];
    const int tx = threadIdx.x & 63;
    const int ty = threadIdx.x >> 6;
    const int n0 = blockIdx.x * 64;
    const int k0 = blockIdx.y * 64;
#pragma unroll
    for (int i = 0; i < 16; ++i) {
        int rk = ty + 4 * i;
        tile[tx][rk] = (_Float16)W[(size_t)(k0 + rk) * N + n0 + tx];
    }
    __syncthreads();
#pragma unroll
    for (int i = 0; i < 16; ++i) {
        int rn = ty + 4 * i;
        Wt[(size_t)(n0 + rn) * K + k0 + tx] = tile[rn][tx];
    }
}

// ---------------------------------------------------------------------------
// GEMM: C[M][N] = A[M][K] @ Wt[N][K]^T + bias[N]
//   A_F16: 0 -> A fp32 (convert on load), 1 -> A f16
//   OUT_MODE: 0 f16 row-major | 1 f16 transposed for V | 2 fp32 row-major
// Block 256 thr = 8 waves; each wave computes 32(M) x 64(N).
// ---------------------------------------------------------------------------
template <int A_F16, int OUT_MODE>
__global__ __launch_bounds__(256)
void xattn_gemm_wmma(const void* __restrict__ Av, const _Float16* __restrict__ Wt,
                     const float* __restrict__ bias, void* __restrict__ Cv,
                     int M, int N, int K) {
    const int lane = threadIdx.x & 31;
    const int wave = threadIdx.x >> 5;
    const int half = lane >> 4;
    const int r    = lane & 15;
    const int m0   = blockIdx.y * 256 + wave * 32;
    const int n0   = blockIdx.x * 64;

    v8f acc[2][4];
#pragma unroll
    for (int mi = 0; mi < 2; ++mi)
#pragma unroll
        for (int j = 0; j < 4; ++j) acc[mi][j] = (v8f)(0.0f);

    const float*    Af = (const float*)Av;
    const _Float16* Ah = (const _Float16*)Av;

    for (int k0 = 0; k0 < K; k0 += 32) {
        v16h a[2];
#pragma unroll
        for (int mi = 0; mi < 2; ++mi) {
            int row = m0 + 16 * mi + r;
            if (A_F16) {
                a[mi] = frag_h16(Ah + (size_t)row * K + k0, half);
                if (k0 + 64 < K) __builtin_prefetch(Ah + (size_t)row * K + k0 + 64, 0, 1);
            } else {
                a[mi] = frag_f32(Af + (size_t)row * K + k0, half);
                if (k0 + 64 < K) __builtin_prefetch(Af + (size_t)row * K + k0 + 64, 0, 1);
            }
        }
        v16h b[4];
#pragma unroll
        for (int j = 0; j < 4; ++j)
            b[j] = frag_h16(Wt + (size_t)(n0 + 16 * j + r) * K + k0, half);
#pragma unroll
        for (int mi = 0; mi < 2; ++mi)
#pragma unroll
            for (int j = 0; j < 4; ++j)
                acc[mi][j] = wmma_f16(a[mi], b[j], acc[mi][j]);
    }

#pragma unroll
    for (int mi = 0; mi < 2; ++mi) {
#pragma unroll
        for (int j = 0; j < 4; ++j) {
            const int n     = n0 + 16 * j + r;
            const int mbase = m0 + 16 * mi + 8 * half;
            const float bs  = bias[n];
            if (OUT_MODE == 0) {
                _Float16* C = (_Float16*)Cv;
#pragma unroll
                for (int jj = 0; jj < 8; ++jj)
                    C[(size_t)(mbase + jj) * N + n] = (_Float16)(acc[mi][j][jj] + bs);
            } else if (OUT_MODE == 2) {
                float* C = (float*)Cv;
#pragma unroll
                for (int jj = 0; jj < 8; ++jj)
                    C[(size_t)(mbase + jj) * N + n] = acc[mi][j][jj] + bs;
            } else {
                const int bi  = mbase >> 10;          // SKV == 1024
                const int skv = mbase & (SKV - 1);
                const int hh  = n >> 6;               // DHEAD == 64
                const int dd  = n & (DHEAD - 1);
                _Float16* C = (_Float16*)Cv +
                              ((size_t)((bi * HEADS + hh) * DHEAD + dd)) * SKV + skv;
                v8h v;
#pragma unroll
                for (int jj = 0; jj < 8; ++jj) v[jj] = (_Float16)(acc[mi][j][jj] + bs);
                *(v8h*)C = v;
            }
        }
    }
}

// ---------------------------------------------------------------------------
// Flash attention, TDM-staged.  WG = 8 waves, all on the same (b,h), covering
// 128 query rows.  Per 64-kv chunk, wave 0 issues two tensor_load_to_lds
// (K 64x64, V^T 64x64, HW-padded rows), double-buffered and overlapped with
// compute; TENSORcnt + workgroup barrier synchronize.  Each wave handles 16
// queries via S^T = K.Q^T  /  O^T = V^T.P^T  (all-in-lane online softmax).
// ---------------------------------------------------------------------------
__global__ __launch_bounds__(256)
void xattn_flash_attn(const _Float16* __restrict__ Q,   // [B][SQ][EMB] f16
                      const _Float16* __restrict__ Kc,  // [B][SKV][EMB] f16
                      const _Float16* __restrict__ Vt,  // [B][H][DHEAD][SKV] f16
                      _Float16* __restrict__ O) {       // [B][SQ][EMB] f16
    __shared__ _Float16 Ks[2][64 * LDSROW];
    __shared__ _Float16 Vs[2][64 * LDSROW];

    const int lane = threadIdx.x & 31;
    const int wave = threadIdx.x >> 5;
    const int half = lane >> 4;
    const int r    = lane & 15;
    const int h    = blockIdx.y;
    const int b    = blockIdx.z;
    const int q0   = blockIdx.x * 128 + wave * 16;

    const _Float16* Qp = Q  + ((size_t)(b * SQ + q0)) * EMB + h * DHEAD;
    const _Float16* Kp = Kc + ((size_t)(b * SKV)) * EMB + h * DHEAD;
    const _Float16* Vp = Vt + ((size_t)(b * HEADS + h) * DHEAD) * SKV;

    // Q fragments (B operand of S^T), reused across all kv chunks
    const v16h fq0 = frag_h16(Qp + (size_t)r * EMB + 0,  half);
    const v16h fq1 = frag_h16(Qp + (size_t)r * EMB + 32, half);

    v8f o[4];
#pragma unroll
    for (int t = 0; t < 4; ++t) o[t] = (v8f)(0.0f);

    float m_run = -3.0e38f;
    float l_run = 0.0f;
    const float scale = 0.125f;     // 1/sqrt(64)

    // prologue: stage chunk 0
    if (wave == 0) {
        tdm_load_2d_f16((unsigned int)(__UINTPTR_TYPE__)&Ks[0][0], Kp, 64, 64, EMB);
        tdm_load_2d_f16((unsigned int)(__UINTPTR_TYPE__)&Vs[0][0], Vp, 64, 64, SKV);
        __builtin_amdgcn_s_wait_tensorcnt(0);
    }
    __syncthreads();

    const int NSTEP = SKV / 64;
    for (int step = 0; step < NSTEP; ++step) {
        const int cur = step & 1;
        if (wave == 0 && step + 1 < NSTEP) {      // async-prefetch next chunk
            tdm_load_2d_f16((unsigned int)(__UINTPTR_TYPE__)&Ks[cur ^ 1][0],
                            Kp + (size_t)(step + 1) * 64 * EMB, 64, 64, EMB);
            tdm_load_2d_f16((unsigned int)(__UINTPTR_TYPE__)&Vs[cur ^ 1][0],
                            Vp + (step + 1) * 64, 64, 64, SKV);
        }
        const _Float16* Kl = Ks[cur];
        const _Float16* Vl = Vs[cur];

#pragma unroll
        for (int sub = 0; sub < 2; ++sub) {       // two 32-kv substeps
            const int kvl = sub * 32;
            // ---- S^T tiles (16 kv x 16 q each) from LDS
            v8f s0 = (v8f)(0.0f), s1 = (v8f)(0.0f);
            v16h ka;
            ka = frag_h16(Kl + (kvl + r) * LDSROW + 0,  half); s0 = wmma_f16(ka, fq0, s0);
            ka = frag_h16(Kl + (kvl + r) * LDSROW + 32, half); s0 = wmma_f16(ka, fq1, s0);
            ka = frag_h16(Kl + (kvl + 16 + r) * LDSROW + 0,  half); s1 = wmma_f16(ka, fq0, s1);
            ka = frag_h16(Kl + (kvl + 16 + r) * LDSROW + 32, half); s1 = wmma_f16(ka, fq1, s1);

            // lane owns q = q0+r; its 16 S values cover kv = {8h+j, 16+8h+j}
            float sv[16];
            float smax = -3.0e38f;
#pragma unroll
            for (int j = 0; j < 8; ++j) {
                sv[j]     = s0[j] * scale;
                sv[8 + j] = s1[j] * scale;
            }
#pragma unroll
            for (int t = 0; t < 16; ++t) smax = fmaxf(smax, sv[t]);
            smax = fmaxf(smax, __shfl_xor(smax, 16));

            const float m_new = fmaxf(m_run, smax);
            const float corr  = __expf(m_run - m_new);
            float lsum = 0.0f;
#pragma unroll
            for (int t = 0; t < 16; ++t) { sv[t] = __expf(sv[t] - m_new); lsum += sv[t]; }
            lsum += __shfl_xor(lsum, 16);
            l_run = l_run * corr + lsum;
            m_run = m_new;

#pragma unroll
            for (int t = 0; t < 4; ++t)
#pragma unroll
                for (int jj = 0; jj < 8; ++jj) o[t][jj] *= corr;

            // P^T B-fragment: bit-compatible with S^T accumulator layout
            v16h pb;
#pragma unroll
            for (int j = 0; j < 8; ++j) {
                pb[j]     = (_Float16)sv[j];
                pb[8 + j] = (_Float16)sv[8 + j];
            }

            // O^T += V^T(d-tile, kv-chunk) . P^T
#pragma unroll
            for (int t = 0; t < 4; ++t) {
                v16h va = frag_h16(Vl + (16 * t + r) * LDSROW + kvl, half);
                o[t] = wmma_f16(va, pb, o[t]);
            }
        }

        if (wave == 0 && step + 1 < NSTEP) __builtin_amdgcn_s_wait_tensorcnt(0);
        __syncthreads();
    }

    // epilogue: lane q = q0+r ; VGPR jj of tile t -> d = 16t + 8*half + jj
    const float inv_l = 1.0f / l_run;
    _Float16* Op = O + ((size_t)(b * SQ + q0 + r)) * EMB + h * DHEAD;
#pragma unroll
    for (int t = 0; t < 4; ++t) {
        v8h ov;
#pragma unroll
        for (int jj = 0; jj < 8; ++jj) ov[jj] = (_Float16)(o[t][jj] * inv_l);
        *(v8h*)(Op + 16 * t + 8 * half) = ov;
    }
}

// ---------------------------------------------------------------------------
// host side
// ---------------------------------------------------------------------------
extern "C" void kernel_launch(void* const* d_in, const int* in_sizes, int n_in,
                              void* d_out, int out_size, void* d_ws, size_t ws_size,
                              hipStream_t stream) {
    const float* x  = (const float*)d_in[0];
    const float* y  = (const float*)d_in[1];
    const float* Wq = (const float*)d_in[2];
    const float* bq = (const float*)d_in[3];
    const float* Wk = (const float*)d_in[4];
    const float* bk = (const float*)d_in[5];
    const float* Wv = (const float*)d_in[6];
    const float* bv = (const float*)d_in[7];
    const float* Wo = (const float*)d_in[8];
    const float* bo = (const float*)d_in[9];

    char* ws = (char*)d_ws;
    const size_t MQ = (size_t)BATCH * SQ;      // 16384
    const size_t MK = (size_t)BATCH * SKV;     // 4096
    _Float16* Qh   = (_Float16*)(ws);
    _Float16* Kh   = (_Float16*)(ws + MQ * EMB * 2);
    _Float16* Vth  = (_Float16*)(ws + (MQ + MK) * EMB * 2);
    _Float16* Ah   = (_Float16*)(ws + (MQ + 2 * MK) * EMB * 2);
    char*     wbase = ws + (2 * MQ + 2 * MK) * EMB * 2;
    _Float16* Wqt = (_Float16*)(wbase);
    _Float16* Wkt = (_Float16*)(wbase + (size_t)EMB * EMB * 2);
    _Float16* Wvt = (_Float16*)(wbase + (size_t)EMB * EMB * 2 + (size_t)EMB * CDIM * 2);
    _Float16* Wot = (_Float16*)(wbase + (size_t)EMB * EMB * 2 + 2 * (size_t)EMB * CDIM * 2);

    xattn_transpose_convert<<<dim3(EMB / 64, EMB / 64), 256, 0, stream>>>(Wq, Wqt, EMB, EMB);
    xattn_transpose_convert<<<dim3(EMB / 64, CDIM / 64), 256, 0, stream>>>(Wk, Wkt, CDIM, EMB);
    xattn_transpose_convert<<<dim3(EMB / 64, CDIM / 64), 256, 0, stream>>>(Wv, Wvt, CDIM, EMB);
    xattn_transpose_convert<<<dim3(EMB / 64, EMB / 64), 256, 0, stream>>>(Wo, Wot, EMB, EMB);

    xattn_gemm_wmma<0, 0><<<dim3(EMB / 64, MQ / 256), 256, 0, stream>>>(
        x, Wqt, bq, Qh, (int)MQ, EMB, EMB);
    xattn_gemm_wmma<0, 0><<<dim3(EMB / 64, MK / 256), 256, 0, stream>>>(
        y, Wkt, bk, Kh, (int)MK, EMB, CDIM);
    xattn_gemm_wmma<0, 1><<<dim3(EMB / 64, MK / 256), 256, 0, stream>>>(
        y, Wvt, bv, Vth, (int)MK, EMB, CDIM);

    xattn_flash_attn<<<dim3(SQ / 128, HEADS, BATCH), 256, 0, stream>>>(Qh, Kh, Vth, Ah);

    xattn_gemm_wmma<1, 2><<<dim3(EMB / 64, MQ / 256), 256, 0, stream>>>(
        Ah, Wot, bo, (float*)d_out, (int)MQ, EMB, EMB);
}
